// GPTSelfAttn_18554258718947
// MI455X (gfx1250) — matrix operand order repaired and checked by reading, not compiled, not run on hardware
//
#include <hip/hip_runtime.h>
#include <hip/hip_bf16.h>
#include <math.h>

// ---------------- problem constants (from reference) ----------------
#define TDIM   1024
#define BBATCH 2
#define NBT    (BBATCH * TDIM)     // 2048 token rows
#define CDIM   1024
#define HHEADS 16
#define DHEAD  64
#define FFDIM  4096
#define NLAYER 4
#define VVOCAB 32000
#define LNEPS  1e-5f
#define ATT_SCALE 0.125f           // 1/sqrt(64)

typedef __attribute__((ext_vector_type(16))) __bf16 v16bf;
typedef __attribute__((ext_vector_type(8)))  __bf16 v8bf;
typedef __attribute__((ext_vector_type(4)))  __bf16 v4bf;
typedef __attribute__((ext_vector_type(8)))  float  v8f;

// ---- CDNA5 staging-path selection (probe via __has_builtin) ----
#if defined(__has_builtin)
#if __has_builtin(__builtin_amdgcn_tensor_load_to_lds)
#define USE_TDM 1
#if __clang_major__ >= 23
#define TDM_SIX 1
#endif
#endif
#if !defined(USE_TDM) && __has_builtin(__builtin_amdgcn_global_load_async_to_lds_b128)
#define USE_ASYNC 1
#endif
#endif
#ifndef USE_TDM
#define USE_TDM 0
#endif
#ifndef USE_ASYNC
#define USE_ASYNC 0
#endif
#ifndef TDM_SIX
#define TDM_SIX 0
#endif
#define A_F32 (USE_TDM || USE_ASYNC)

#if defined(__AMDGCN__)
#if USE_TDM
#warning "gfx1250 A-staging path: TDM tensor_load_to_lds"
#elif USE_ASYNC
#warning "gfx1250 A-staging path: global_load_async_to_lds_b128"
#else
#warning "gfx1250 A-staging path: register fallback (no TDM/async builtins)"
#endif
#endif

static __device__ __forceinline__ __bf16 f2bf(float f) { return (__bf16)f; }

static __device__ __forceinline__ v8f v8f_zero() {
  v8f z = {0.f, 0.f, 0.f, 0.f, 0.f, 0.f, 0.f, 0.f};
  return z;
}

static __device__ __forceinline__ v8f wmma_bf16(v16bf a, v16bf b, v8f c) {
  // v_wmma_f32_16x16x32_bf16  D = A(16x32) x B(32x16) + C
  return __builtin_amdgcn_wmma_f32_16x16x32_bf16(false, a, false, b, (short)0, c,
                                                 false, false);
}

#if USE_TDM
typedef unsigned int v4u __attribute__((ext_vector_type(4)));
typedef int          v8i __attribute__((ext_vector_type(8)));
typedef int          v4i __attribute__((ext_vector_type(4)));

static __device__ __forceinline__ void wait_tensorcnt0() {
#if defined(__has_builtin) && __has_builtin(__builtin_amdgcn_s_wait_tensorcnt)
  __builtin_amdgcn_s_wait_tensorcnt(0);
#else
  asm volatile("s_wait_tensorcnt 0x0" ::: "memory");
#endif
}

// Low 32 bits of a generic LDS pointer = LDS byte address (ISA 10.2 aperture map).
static __device__ __forceinline__ unsigned lds_off(const void* p) {
  return (unsigned)(size_t)p;
}

// One 2-D fp32 tile, global -> LDS, LDS row padding 4 dwords per 32 (stride 36).
static __device__ __forceinline__ void tdm_load_2d_f32(const float* gsrc,
                                                       unsigned lds_byte,
                                                       unsigned tensor_w,
                                                       unsigned tensor_h,
                                                       unsigned row_stride_elems,
                                                       unsigned tile_w,
                                                       unsigned tile_h) {
  unsigned long long ga = (unsigned long long)(size_t)(const void*)gsrc;
  v4u g0;
  g0[0] = 1u;                                             // count=1 user D#
  g0[1] = lds_byte;                                       // lds_addr
  g0[2] = (unsigned)(ga & 0xffffffffu);                   // global_addr[31:0]
  g0[3] = (unsigned)((ga >> 32) & 0x01ffffffu) | 0x80000000u; // addr[56:32]|type=2
  v8i g1;
  g1[0] = (2 << 16) | (1 << 20) | (4 << 22) | (3 << 25);  // 4B elems, pad 4dw/32dw
  g1[1] = (int)((tensor_w & 0xffffu) << 16);              // dim0 lo16 (atomic_addr=0)
  g1[2] = (int)((tensor_w >> 16) | ((tensor_h & 0xffffu) << 16)); // dim0 hi|dim1 lo
  g1[3] = (int)((tensor_h >> 16) | (tile_w << 16));       // dim1 hi | tile_dim0
  g1[4] = (int)(tile_h & 0xffffu);                        // tile_dim1 (tile_dim2=0)
  g1[5] = (int)row_stride_elems;                          // dim0_stride lo32
  g1[6] = 0;
  g1[7] = 0;
  v4i z = {0, 0, 0, 0};
#if TDM_SIX
  v8i z8 = {0, 0, 0, 0, 0, 0, 0, 0};
  __builtin_amdgcn_tensor_load_to_lds(g0, g1, z, z, z8, 0);
#else
  __builtin_amdgcn_tensor_load_to_lds(g0, g1, z, z, 0);
#endif
}
#endif  // USE_TDM

#if USE_ASYNC
static __device__ __forceinline__ void async_cp_b128(const float* g, float* l) {
  __builtin_amdgcn_global_load_async_to_lds_b128(
      (const __attribute__((address_space(1))) void*)g,
      (__attribute__((address_space(3))) void*)l, 0, 0);
}
static __device__ __forceinline__ void wait_async0() {
#if defined(__has_builtin) && __has_builtin(__builtin_amdgcn_s_wait_asynccnt)
  __builtin_amdgcn_s_wait_asynccnt(0);
#else
  asm volatile("s_wait_asynccnt 0x0" ::: "memory");
#endif
}
#endif  // USE_ASYNC

// -------- fragment loaders (ISA 7.12.2 layouts, wave32) --------
// A 16x32 bf16 frag from global fp32: lane = khalf*16 + m;
//   elems 0..7  -> K = khalf*8 + 0..7 ; elems 8..15 -> K = 16 + khalf*8 + 0..7
static __device__ __forceinline__ v16bf load_a_frag_f32(const float* __restrict__ A,
                                                        int lda, int row0, int k0,
                                                        int lane) {
  int m = lane & 15, half = lane >> 4;
  const float* r = A + (size_t)(row0 + m) * lda + k0 + half * 8;
  v16bf a;
#pragma unroll
  for (int v = 0; v < 4; ++v) {
    float2 p = *(const float2*)(r + v * 2);
    a[2 * v]     = f2bf(p.x);
    a[2 * v + 1] = f2bf(p.y);
  }
  const float* r2 = r + 16;
#pragma unroll
  for (int v = 0; v < 4; ++v) {
    float2 p = *(const float2*)(r2 + v * 2);
    a[8 + 2 * v]     = f2bf(p.x);
    a[8 + 2 * v + 1] = f2bf(p.y);
  }
  return a;
}

// A frag from an fp32 LDS tile, row stride S dwords (S*4 % 16 == 0): 4 aligned b128.
static __device__ __forceinline__ v16bf frag_a_lds_f32(const float* T, int S, int lane) {
  int m = lane & 15, khalf = lane >> 4;
  const float* r = T + m * S + khalf * 8;
  float4 p0 = *(const float4*)(r);
  float4 p1 = *(const float4*)(r + 4);
  float4 p2 = *(const float4*)(r + 16);
  float4 p3 = *(const float4*)(r + 20);
  v16bf a;
  a[0] = f2bf(p0.x);  a[1] = f2bf(p0.y);  a[2] = f2bf(p0.z);  a[3] = f2bf(p0.w);
  a[4] = f2bf(p1.x);  a[5] = f2bf(p1.y);  a[6] = f2bf(p1.z);  a[7] = f2bf(p1.w);
  a[8] = f2bf(p2.x);  a[9] = f2bf(p2.y);  a[10] = f2bf(p2.z); a[11] = f2bf(p2.w);
  a[12] = f2bf(p3.x); a[13] = f2bf(p3.y); a[14] = f2bf(p3.z); a[15] = f2bf(p3.w);
  return a;
}

// A frag from a bf16 LDS tile, row stride S bf16 elems (S*2 % 16 == 0): 2 b128.
static __device__ __forceinline__ v16bf frag_a_lds(const __bf16* T, int S, int lane) {
  int m = lane & 15, khalf = lane >> 4;
  const __bf16* r = T + m * S + khalf * 8;
  union { v16bf f; v8bf h[2]; } u;
  u.h[0] = *(const v8bf*)(r);
  u.h[1] = *(const v8bf*)(r + 16);
  return u.f;
}

// B 32x16 frag from an N-major (transposed) bf16 LDS tile: row = n, cols = K.
// Lane needs K = koff + khalf*8 + 0..7 and koff + 16 + khalf*8 + 0..7 -> two aligned 16B ds reads.
static __device__ __forceinline__ v16bf frag_bT_lds(const __bf16* T, int S, int koff,
                                                    int lane) {
  int n = lane & 15, khalf = lane >> 4;
  const __bf16* r = T + n * S + koff + khalf * 8;
  union { v16bf f; v8bf h[2]; } u;
  u.h[0] = *(const v8bf*)(r);
  u.h[1] = *(const v8bf*)(r + 16);
  return u.f;
}

// ---------------- embedding gather ----------------
__global__ void gpt_embed(const int* __restrict__ idx, const float* __restrict__ tok,
                          float* __restrict__ x) {
  int gid = blockIdx.x * blockDim.x + threadIdx.x;       // over NBT*CDIM
  int bt = gid >> 10;                                    // CDIM = 1024
  int c  = gid & (CDIM - 1);
  x[(size_t)bt * CDIM + c] = tok[(size_t)idx[bt] * CDIM + c];
}

// ---------------- layernorm (one block per row) ----------------
__global__ void gpt_layernorm(const float* __restrict__ x, const float* __restrict__ g,
                              const float* __restrict__ b, float* __restrict__ out) {
  __shared__ float red[256];
  int row = blockIdx.x;
  const float* xr = x + (size_t)row * CDIM;
  float v0[4], s = 0.f;
#pragma unroll
  for (int i = 0; i < 4; ++i) {
    v0[i] = xr[threadIdx.x + i * 256];
    s += v0[i];
  }
  red[threadIdx.x] = s;
  __syncthreads();
  for (int o = 128; o > 0; o >>= 1) {
    if (threadIdx.x < o) red[threadIdx.x] += red[threadIdx.x + o];
    __syncthreads();
  }
  float mu = red[0] * (1.f / CDIM);
  __syncthreads();
  float sq = 0.f;
#pragma unroll
  for (int i = 0; i < 4; ++i) {
    float d = v0[i] - mu;
    sq += d * d;
  }
  red[threadIdx.x] = sq;
  __syncthreads();
  for (int o = 128; o > 0; o >>= 1) {
    if (threadIdx.x < o) red[threadIdx.x] += red[threadIdx.x + o];
    __syncthreads();
  }
  float rstd = rsqrtf(red[0] * (1.f / CDIM) + LNEPS);
  float* orow = out + (size_t)row * CDIM;
#pragma unroll
  for (int i = 0; i < 4; ++i) {
    int c = threadIdx.x + i * 256;
    orow[c] = (v0[i] - mu) * rstd * g[c] + b[c];
  }
}

// ---------------- RoPE (out of place; even/odd -> halves) ----------------
__global__ void gpt_rope(const float* __restrict__ in, float* __restrict__ out) {
  int gid = blockIdx.x * blockDim.x + threadIdx.x;  // over NBT*HHEADS*(DHEAD/2)
  int i  = gid & 31;
  int h  = (gid >> 5) & (HHEADS - 1);
  int bt = gid >> 9;
  int t  = bt & (TDIM - 1);
  float inv = __expf(-(float)(2 * i) * (logf(10000.0f) / (float)DHEAD));
  float ang = (float)t * inv;
  float cs, sn;
  __sincosf(ang, &sn, &cs);
  const float* r = in + (size_t)bt * CDIM + h * DHEAD;
  float e = r[2 * i], o = r[2 * i + 1];
  float* w = out + (size_t)bt * CDIM + h * DHEAD;
  w[i]      = e * cs - o * sn;
  w[i + 32] = e * sn + o * cs;
}

// ---------------- LDS-staged WMMA GEMM with fused epilogue ----------------
// Out(MxN) = A(MxK) @ B(KxN) [+bias] [gelu] [+residual]
// block = 256 threads (8 waves). Tile: 128 rows x 64 cols, K-step 32.
// A tile: TDM/async paths stage raw fp32 (36-dword padded rows); register
// fallback stages bf16 (40-elem rows). B tile: bf16, transposed/N-major.
#define MBLK   128
#define NBLK   64
#define KSTEP  32
#define LDA_F  36   // fp32 elems per A row in LDS (32 + 4 pad) = 144B
#define LDA_S  40   // bf16 elems per A row in LDS (32 + 8 pad) = 80B
#define LDB_S  40   // bf16 elems per B^T row (32 + 8 pad) = 80B

__global__ __launch_bounds__(256, 1) void gpt_gemm(const float* __restrict__ A,
                                                   const float* __restrict__ B,
                                                   const float* __restrict__ bias,
                                                   const float* __restrict__ residual,
                                                   float* __restrict__ Out, int M,
                                                   int N, int K, int applyGelu) {
#if A_F32
  __shared__ float  Asm[2][MBLK * LDA_F];   // 2 x 18 KB
#else
  __shared__ __bf16 Asm[2][MBLK * LDA_S];   // 2 x 10 KB
#endif
  __shared__ __bf16 Bsm[2][NBLK * LDB_S];   // 2 x 5 KB
  int tid = threadIdx.x, lane = tid & 31, wid = tid >> 5;
  int row0 = blockIdx.y * MBLK;
  int col0 = blockIdx.x * NBLK;

#if A_F32
  // Dynamically dead store: keeps LLVM from undef-folding Asm loads when no
  // source-level store writes Asm (TDM / async staging).
  if (M == -1) { Asm[0][tid] = 0.f; Asm[1][tid] = 0.f; }
#endif

  float4 bR[2];
#if !A_F32
  float4 aR[4];
#endif

  auto loadB = [&](int k0) {
#pragma unroll
    for (int i = 0; i < 2; ++i) {            // B: 32x64 f32, 2 float4/thread
      int idx4 = tid + 256 * i;
      int kk = idx4 >> 4, c = (idx4 & 15) * 4;
      bR[i] = *(const float4*)(B + (size_t)(k0 + kk) * N + col0 + c);
    }
  };
  auto storeB = [&](int s) {
#pragma unroll
    for (int i = 0; i < 2; ++i) {            // transpose into N-major bf16
      int idx4 = tid + 256 * i;
      int kk = idx4 >> 4, c = (idx4 & 15) * 4;
      Bsm[s][(c + 0) * LDB_S + kk] = f2bf(bR[i].x);
      Bsm[s][(c + 1) * LDB_S + kk] = f2bf(bR[i].y);
      Bsm[s][(c + 2) * LDB_S + kk] = f2bf(bR[i].z);
      Bsm[s][(c + 3) * LDB_S + kk] = f2bf(bR[i].w);
    }
  };
#if USE_TDM
  auto stageA = [&](int s, int k0) {         // one descriptor per K-step
    if (wid == 0)
      tdm_load_2d_f32(A + (size_t)row0 * K + k0, lds_off(&Asm[s][0]),
                      (unsigned)K, 0x100000u, (unsigned)K, KSTEP, MBLK);
  };
#elif USE_ASYNC
  auto stageA = [&](int s, int k0) {         // per-lane async b128, padded dest
#pragma unroll
    for (int i = 0; i < 4; ++i) {
      int idx4 = tid + 256 * i;
      int r = idx4 >> 3, c = (idx4 & 7) * 4;
      async_cp_b128(A + (size_t)(row0 + r) * K + k0 + c, &Asm[s][r * LDA_F + c]);
    }
  };
#else
  auto loadA = [&](int k0) {
#pragma unroll
    for (int i = 0; i < 4; ++i) {            // A: 128x32 f32, 4 float4/thread
      int idx4 = tid + 256 * i;
      int r = idx4 >> 3, c = (idx4 & 7) * 4;
      aR[i] = *(const float4*)(A + (size_t)(row0 + r) * K + k0 + c);
    }
  };
  auto storeA = [&](int s) {
#pragma unroll
    for (int i = 0; i < 4; ++i) {
      int idx4 = tid + 256 * i;
      int r = idx4 >> 3, c = (idx4 & 7) * 4;
      v4bf pk = {f2bf(aR[i].x), f2bf(aR[i].y), f2bf(aR[i].z), f2bf(aR[i].w)};
      *(v4bf*)&Asm[s][r * LDA_S + c] = pk;   // 8B aligned ds_store_b64
    }
  };
#endif

  auto stage_wait = [&]() {
#if USE_TDM
    if (wid == 0) wait_tensorcnt0();
#elif USE_ASYNC
    wait_async0();
#endif
  };

  v8f acc[4] = {v8f_zero(), v8f_zero(), v8f_zero(), v8f_zero()};

#if A_F32
  stageA(0, 0);
#else
  loadA(0);
  storeA(0);
#endif
  loadB(0);
  storeB(0);
  stage_wait();
  __syncthreads();

  int s = 0;
  for (int k0 = 0; k0 < K; k0 += KSTEP) {
    int kn = k0 + KSTEP;
    if (kn < K) {
#if A_F32
      stageA(s ^ 1, kn);
#else
      loadA(kn);
#endif
      loadB(kn);
      if (kn + KSTEP < K) {                   // K+2 tile: global_prefetch_b8
        __builtin_prefetch(A + (size_t)(row0 + (tid >> 3)) * K + kn + KSTEP, 0, 1);
        __builtin_prefetch(B + (size_t)(kn + KSTEP + (tid >> 4)) * N + col0, 0, 1);
      }
    }
#if A_F32
    v16bf a = frag_a_lds_f32(&Asm[s][wid * 16 * LDA_F], LDA_F, lane);
#else
    v16bf a = frag_a_lds(&Asm[s][wid * 16 * LDA_S], LDA_S, lane);
#endif
#pragma unroll
    for (int j = 0; j < 4; ++j) {
      v16bf b = frag_bT_lds(&Bsm[s][(j * 16) * LDB_S], LDB_S, 0, lane);
      acc[j] = wmma_bf16(a, b, acc[j]);
    }
    if (kn < K) {
#if !A_F32
      storeA(s ^ 1);
#endif
      storeB(s ^ 1);
      stage_wait();
      __syncthreads();
      s ^= 1;
    }
  }

  int hi8 = (lane >> 4) * 8, nl = lane & 15;
  int wrow0 = row0 + wid * 16;
#pragma unroll
  for (int j = 0; j < 4; ++j) {
    int col = col0 + 16 * j + nl;
    float bb = bias ? bias[col] : 0.f;
#pragma unroll
    for (int v = 0; v < 8; ++v) {
      int row = wrow0 + hi8 + v;
      float val = acc[j][v] + bb;
      if (applyGelu) val = 0.5f * val * (1.0f + erff(val * 0.70710678118654752f));
      if (residual) val += residual[(size_t)row * N + col];
      Out[(size_t)row * N + col] = val;
    }
  }
}

// ---------------- flash attention ----------------
// block = 128 threads (4 waves) sharing one (b, h); wave w owns q tile g*4+w.
// K tile (row-major bf16, stride 80) and V tile (transposed bf16, stride 40)
// cooperatively staged in LDS; all WMMA operand reads are aligned b128 loads.
#define KLDS 80   // 64 + 16 pad bf16 per K-tile row (160B)
#define VLDS 40   // 32 + 8 pad bf16 per V^T row (80B)

__global__ __launch_bounds__(128) void gpt_attn(const float* __restrict__ q,
                                                const float* __restrict__ k,
                                                const float* __restrict__ v,
                                                float* __restrict__ y) {
  __shared__ __bf16 Ksm[32 * KLDS];        // 5 KB
  __shared__ __bf16 Vsm[64 * VLDS];        // 5 KB
  __shared__ __bf16 Pbuf[4][16 * 32];      // 4 KB (per-wave P tiles)
  int tid = threadIdx.x, lane = tid & 31, wid = tid >> 5;
  int bid = blockIdx.x;                    // 512 = 2 * 16 * 16
  int g  = bid & 15;
  int hh = (bid >> 4) & (HHEADS - 1);
  int bb = bid >> 8;
  int qt = g * 4 + wid;
  __bf16* Pw = Pbuf[wid];

  const float* Qb = q + (size_t)bb * TDIM * CDIM + hh * DHEAD;  // row stride CDIM
  const float* Kb = k + (size_t)bb * TDIM * CDIM + hh * DHEAD;
  const float* Vb = v + (size_t)bb * TDIM * CDIM + hh * DHEAD;

  int qrow0 = qt * 16;
  v16bf aq0 = load_a_frag_f32(Qb, CDIM, qrow0, 0, lane);
  v16bf aq1 = load_a_frag_f32(Qb, CDIM, qrow0, 32, lane);

  int hi8 = (lane >> 4) * 8, nl = lane & 15;
  float m_i[8], s_i[8];
  v8f o[4] = {v8f_zero(), v8f_zero(), v8f_zero(), v8f_zero()};
#pragma unroll
  for (int r = 0; r < 8; ++r) { m_i[r] = -1e30f; s_i[r] = 0.f; }

  int kmaxMe  = (qt + 1) * 16;             // this wave's causal key bound
  int kmaxBlk = (g * 4 + 4) * 16;          // block-wide bound

  for (int kb = 0; kb < kmaxBlk; kb += 32) {
    __syncthreads();                        // previous iter's LDS reads done
    // ---- cooperative staging: K rows kb..kb+31 and V^T ----
#pragma unroll
    for (int i = 0; i < 4; ++i) {           // 2048 f32 each, 4 float4/thread
      int idx4 = tid + 128 * i;
      int r = idx4 >> 4;                    // key 0..31
      int c = (idx4 & 15) * 4;              // d 0..60
      float4 pk4 = *(const float4*)(Kb + (size_t)(kb + r) * CDIM + c);
      v4bf pk = {f2bf(pk4.x), f2bf(pk4.y), f2bf(pk4.z), f2bf(pk4.w)};
      *(v4bf*)&Ksm[r * KLDS + c] = pk;
      float4 pv4 = *(const float4*)(Vb + (size_t)(kb + r) * CDIM + c);
      Vsm[(c + 0) * VLDS + r] = f2bf(pv4.x);
      Vsm[(c + 1) * VLDS + r] = f2bf(pv4.y);
      Vsm[(c + 2) * VLDS + r] = f2bf(pv4.z);
      Vsm[(c + 3) * VLDS + r] = f2bf(pv4.w);
    }
    __syncthreads();
    if (kb >= kmaxMe) continue;             // past causal bound: barriers only

    // ---- scores: S = Q(16x64) @ K^T(64x32), two 16-key subtiles ----
    v8f sc[2];
#pragma unroll
    for (int hsub = 0; hsub < 2; ++hsub) {
      const __bf16* Kt = &Ksm[(hsub * 16) * KLDS];
      v8f t0 = wmma_bf16(aq0, frag_bT_lds(Kt, KLDS, 0, lane), v8f_zero());
      sc[hsub] = wmma_bf16(aq1, frag_bT_lds(Kt, KLDS, 32, lane), t0);
    }
    // ---- online softmax over this 32-key block ----
#pragma unroll
    for (int r = 0; r < 8; ++r) {
      int rowg = qrow0 + hi8 + r;
      float a0 = sc[0][r] * ATT_SCALE;
      float a1 = sc[1][r] * ATT_SCALE;
      if (kb + nl > rowg)      a0 = -1e30f;
      if (kb + 16 + nl > rowg) a1 = -1e30f;
      float rm = fmaxf(a0, a1);
#pragma unroll
      for (int msk = 1; msk < 16; msk <<= 1) rm = fmaxf(rm, __shfl_xor(rm, msk, 32));
      float mnew = fmaxf(m_i[r], rm);
      float p0 = __expf(a0 - mnew);
      float p1 = __expf(a1 - mnew);
      float rs = p0 + p1;
#pragma unroll
      for (int msk = 1; msk < 16; msk <<= 1) rs += __shfl_xor(rs, msk, 32);
      float corr = __expf(m_i[r] - mnew);
      s_i[r] = s_i[r] * corr + rs;
      m_i[r] = mnew;
#pragma unroll
      for (int j = 0; j < 4; ++j) o[j][r] *= corr;
      Pw[(hi8 + r) * 32 + nl]      = f2bf(p0);
      Pw[(hi8 + r) * 32 + 16 + nl] = f2bf(p1);
    }
    // ---- O += P(16x32) @ V(32x64) ----
    v16bf ap = frag_a_lds(Pw, 32, lane);
#pragma unroll
    for (int j = 0; j < 4; ++j) {
      v16bf bv = frag_bT_lds(&Vsm[(j * 16) * VLDS], VLDS, 0, lane);
      o[j] = wmma_bf16(ap, bv, o[j]);
    }
  }
  // ---- normalize and store ----
  float* Yb = y + (size_t)bb * TDIM * CDIM + hh * DHEAD;
#pragma unroll
  for (int j = 0; j < 4; ++j) {
#pragma unroll
    for (int r = 0; r < 8; ++r) {
      int row = qrow0 + hi8 + r;
      Yb[(size_t)row * CDIM + j * 16 + nl] = o[j][r] / s_i[r];
    }
  }
}

// ---------------- host driver ----------------
static inline void launch_gemm(const float* A, const float* B, const float* bias,
                               const float* resid, float* Out, int M, int N, int K,
                               int gelu, hipStream_t s) {
  dim3 grid(N / NBLK, M / MBLK);
  gpt_gemm<<<grid, 256, 0, s>>>(A, B, bias, resid, Out, M, N, K, gelu);
}

extern "C" void kernel_launch(void* const* d_in, const int* in_sizes, int n_in,
                              void* d_out, int out_size, void* d_ws, size_t ws_size,
                              hipStream_t stream) {
  const int*   idx  = (const int*)d_in[0];
  const float* tok  = (const float*)d_in[1];
  const float* Wq   = (const float*)d_in[2];
  const float* bq   = (const float*)d_in[3];
  const float* Wk   = (const float*)d_in[4];
  const float* bk   = (const float*)d_in[5];
  const float* Wv   = (const float*)d_in[6];
  const float* bv   = (const float*)d_in[7];
  const float* Wo   = (const float*)d_in[8];
  const float* bo   = (const float*)d_in[9];
  const float* ln1g = (const float*)d_in[10];
  const float* ln1b = (const float*)d_in[11];
  const float* ln2g = (const float*)d_in[12];
  const float* ln2b = (const float*)d_in[13];
  const float* W1   = (const float*)d_in[14];
  const float* b1   = (const float*)d_in[15];
  const float* W2   = (const float*)d_in[16];
  const float* b2   = (const float*)d_in[17];
  const float* lnfg = (const float*)d_in[18];
  const float* lnfb = (const float*)d_in[19];
  const float* lmh  = (const float*)d_in[20];
  float* out = (float*)d_out;

  const size_t XC = (size_t)NBT * CDIM;
  float* x  = (float*)d_ws;        // NBT*C residual stream
  float* h  = x  + XC;             // LN output
  float* q  = h  + XC;
  float* k  = q  + XC;
  float* v  = k  + XC;
  float* qr = v  + XC;             // rope'd q
  float* kr = qr + XC;             // rope'd k
  float* y  = kr + XC;             // attention output
  float* ff = y  + XC;             // NBT*FF

  gpt_embed<<<(NBT * CDIM) / 256, 256, 0, stream>>>(idx, tok, x);

  for (int l = 0; l < NLAYER; ++l) {
    const size_t wCC = (size_t)l * CDIM * CDIM;
    const size_t wCF = (size_t)l * CDIM * FFDIM;

    gpt_layernorm<<<NBT, 256, 0, stream>>>(x, ln1g + l * CDIM, ln1b + l * CDIM, h);
    launch_gemm(h, Wq + wCC, bq + l * CDIM, nullptr, q, NBT, CDIM, CDIM, 0, stream);
    launch_gemm(h, Wk + wCC, bk + l * CDIM, nullptr, k, NBT, CDIM, CDIM, 0, stream);
    launch_gemm(h, Wv + wCC, bv + l * CDIM, nullptr, v, NBT, CDIM, CDIM, 0, stream);

    int ropeN = NBT * HHEADS * (DHEAD / 2);
    gpt_rope<<<ropeN / 256, 256, 0, stream>>>(q, qr);
    gpt_rope<<<ropeN / 256, 256, 0, stream>>>(k, kr);

    gpt_attn<<<(BBATCH * HHEADS * (TDIM / 16)) / 4, 128, 0, stream>>>(qr, kr, v, y);

    launch_gemm(y, Wo + wCC, bo + l * CDIM, x, x, NBT, CDIM, CDIM, 0, stream);

    gpt_layernorm<<<NBT, 256, 0, stream>>>(x, ln2g + l * CDIM, ln2b + l * CDIM, h);
    launch_gemm(h, W1 + wCF, b1 + l * FFDIM, nullptr, ff, NBT, FFDIM, CDIM, 1, stream);
    launch_gemm(ff, W2 + wCF, b2 + l * CDIM, x, x, NBT, CDIM, FFDIM, 0, stream);
  }

  gpt_layernorm<<<NBT, 256, 0, stream>>>(x, lnfg, lnfb, h);
  launch_gemm(h, lmh, nullptr, nullptr, out, NBT, VVOCAB, CDIM, 0, stream);
}
